// M3Care_6674379178081
// MI455X (gfx1250) — matrix operand order repaired and checked by reading, not compiled, not run on hardware
//
#include <hip/hip_runtime.h>
#include <math.h>

// ---------------------------------------------------------------------------
// M3Care forward for MI455X (gfx1250, wave32). All GEMMs run on the WMMA path
// via __builtin_amdgcn_wmma_f32_16x16x32_bf16 (bf16 inputs, f32 accumulate).
// Operand convention: C = A * B^T, A/B row-major [rows, K] bf16 (weights are
// pre-transposed once). Each wave computes a 32x32 output tile (2x2 WMMA
// register blocking): 4 WMMAs per K-step from 2 A- + 2 B-fragments, doubling
// the WMMA:bytes ratio vs one-tile-per-wave. global_prefetch is issued one
// K-step ahead. Workspace requirement: ~232 MB. Param input order = jax
// pytree flatten (sorted dict keys) after x0,x1,x2,x3,missing.
// ---------------------------------------------------------------------------

typedef __attribute__((ext_vector_type(16))) __bf16 v16bf;
typedef __attribute__((ext_vector_type(8)))  __bf16 v8bf;
typedef __attribute__((ext_vector_type(8)))  float  v8f;

#define DEV __device__ __forceinline__

constexpr int   NB   = 4096;    // batch
constexpr int   HDm  = 256;     // hidden
constexpr float EPSN = 1e-5f;

DEV unsigned short f2bf(float f) {              // round-to-nearest-even f32->bf16
  unsigned u = __float_as_uint(f);
  unsigned r = u + 0x7fffu + ((u >> 16) & 1u);
  return (unsigned short)(r >> 16);
}
DEV float bf2f(unsigned short b) { return __uint_as_float(((unsigned)b) << 16); }

// Per-lane fragment loads: element s of lane half h is k = 16h + s, same order for A and B.
DEV v16bf frag_a(const unsigned short* Ap, int kk, int hi) {
  v8bf a0 = *reinterpret_cast<const v8bf*>(Ap + kk +     (hi ? 16 : 0));
  v8bf a1 = *reinterpret_cast<const v8bf*>(Ap + kk + 8 + (hi ? 16 : 0));
  return __builtin_shufflevector(a0, a1, 0,1,2,3,4,5,6,7,8,9,10,11,12,13,14,15);
}
DEV v16bf frag_b(const unsigned short* Bp, int kk, int hi) {
  return *reinterpret_cast<const v16bf*>(Bp + kk + (hi ? 16 : 0));
}
DEV v8f wmma(v16bf a, v16bf b, v8f c) {
  return __builtin_amdgcn_wmma_f32_16x16x32_bf16(false, a, false, b, (short)0, c, false, false);
}

// ---------------------------------------------------------------------------
// WMMA GEMM: C[Md,Nd] = A[Md,K](bf16) * Bt[Nd,K](bf16)^T. One wave per 32x32
// tile (2x2 WMMA blocking), 8 waves per block, tn fastest so waves in a block
// share A rows. Epilogue: optional bias/ReLU, writes f32 / bf16 / bf16^T.
// ---------------------------------------------------------------------------
template<bool BIAS, bool RELU, bool WF, bool WB, bool WT>
__global__ __launch_bounds__(256) void k_gemm(
    const unsigned short* __restrict__ A, const unsigned short* __restrict__ Bt,
    const float* __restrict__ bias, float* __restrict__ Cf,
    unsigned short* __restrict__ Cb, unsigned short* __restrict__ Ct,
    int Md, int Nd, int K)
{
  const int lane = threadIdx.x & 31;
  const int wave = threadIdx.x >> 5;
  const int ntn  = Nd >> 5;                      // 32-wide tiles
  const long tile = (long)blockIdx.x * (blockDim.x >> 5) + wave;
  const long tot  = (long)(Md >> 5) * ntn;
  if (tile >= tot) return;                       // wave-uniform branch
  const int tm = (int)(tile / ntn), tn = (int)(tile % ntn);
  const int hi = lane >> 4, l16 = lane & 15;
  const unsigned short* Ap0 = A  + (size_t)(tm * 32 + l16) * K;
  const unsigned short* Ap1 = Ap0 + (size_t)16 * K;
  const unsigned short* Bp0 = Bt + (size_t)(tn * 32 + l16) * K;
  const unsigned short* Bp1 = Bp0 + (size_t)16 * K;
  v8f c00 = {0.f,0.f,0.f,0.f,0.f,0.f,0.f,0.f};
  v8f c01 = c00, c10 = c00, c11 = c00;
  for (int kk = 0; kk < K; kk += 32) {
    __builtin_prefetch(Ap0 + kk + 32, 0, 1);
    __builtin_prefetch(Bp0 + kk + 32, 0, 1);
    v16bf a0 = frag_a(Ap0, kk, hi);
    v16bf a1 = frag_a(Ap1, kk, hi);
    v16bf b0 = frag_b(Bp0, kk, hi);
    v16bf b1 = frag_b(Bp1, kk, hi);
    c00 = wmma(a0, b0, c00);
    c01 = wmma(a0, b1, c01);
    c10 = wmma(a1, b0, c10);
    c11 = wmma(a1, b1, c11);
  }
  const v8f acc[4] = { c00, c01, c10, c11 };
#pragma unroll
  for (int i = 0; i < 2; ++i) {
#pragma unroll
    for (int j = 0; j < 2; ++j) {
      const v8f c = acc[i * 2 + j];
      const int n = tn * 32 + j * 16 + l16;
      const float bv = BIAS ? bias[n] : 0.f;
#pragma unroll
      for (int r = 0; r < 8; ++r) {
        const int m = tm * 32 + i * 16 + r + (hi ? 8 : 0);
        float v = c[r] + bv;
        if (RELU) v = fmaxf(v, 0.f);
        if (WF) Cf[(size_t)m * Nd + n] = v;
        if (WB) Cb[(size_t)m * Nd + n] = f2bf(v);
        if (WT) Ct[(size_t)n * Md + m] = f2bf(v);
      }
    }
  }
}

// ---------------------------------------------------------------------------
// Fused Gram + squared-distance + global sum over a 32x32 tile:
// D[i,j] = max(sq_i+sq_j-2(X X^T)_ij, 0), atomicAdd per-wave sum (bandwidth).
// ---------------------------------------------------------------------------
__global__ __launch_bounds__(256) void k_dist(
    const unsigned short* __restrict__ X, const float* __restrict__ sq,
    float* __restrict__ D, float* __restrict__ sum_slot, int K)
{
  const int lane = threadIdx.x & 31;
  const int wave = threadIdx.x >> 5;
  const int ntn  = NB >> 5;
  const long tile = (long)blockIdx.x * (blockDim.x >> 5) + wave;
  const long tot  = (long)(NB >> 5) * ntn;
  if (tile >= tot) return;
  const int tm = (int)(tile / ntn), tn = (int)(tile % ntn);
  const int hi = lane >> 4, l16 = lane & 15;
  const unsigned short* Ap0 = X + (size_t)(tm * 32 + l16) * K;
  const unsigned short* Ap1 = Ap0 + (size_t)16 * K;
  const unsigned short* Bp0 = X + (size_t)(tn * 32 + l16) * K;
  const unsigned short* Bp1 = Bp0 + (size_t)16 * K;
  v8f c00 = {0.f,0.f,0.f,0.f,0.f,0.f,0.f,0.f};
  v8f c01 = c00, c10 = c00, c11 = c00;
  for (int kk = 0; kk < K; kk += 32) {
    v16bf a0 = frag_a(Ap0, kk, hi);
    v16bf a1 = frag_a(Ap1, kk, hi);
    v16bf b0 = frag_b(Bp0, kk, hi);
    v16bf b1 = frag_b(Bp1, kk, hi);
    c00 = wmma(a0, b0, c00);
    c01 = wmma(a0, b1, c01);
    c10 = wmma(a1, b0, c10);
    c11 = wmma(a1, b1, c11);
  }
  const v8f acc[4] = { c00, c01, c10, c11 };
  float local = 0.f;
#pragma unroll
  for (int i = 0; i < 2; ++i) {
#pragma unroll
    for (int j = 0; j < 2; ++j) {
      const v8f c = acc[i * 2 + j];
      const int n = tn * 32 + j * 16 + l16;
      const float sqn = sq[n];
#pragma unroll
      for (int r = 0; r < 8; ++r) {
        const int m = tm * 32 + i * 16 + r + (hi ? 8 : 0);
        float d = fmaxf(sq[m] + sqn - 2.f * c[r], 0.f);
        D[(size_t)m * NB + n] = d;
        local += d;
      }
    }
  }
#pragma unroll
  for (int off = 16; off > 0; off >>= 1) local += __shfl_xor(local, off, 32);
  if (lane == 0) atomicAdd(sum_slot, local);
}

// --------------------------- small utility kernels -------------------------
__global__ void k_cvt(const float* __restrict__ s, unsigned short* __restrict__ d, long n) {
  long i = blockIdx.x * (long)blockDim.x + threadIdx.x;
  if (i < n) d[i] = f2bf(s[i]);
}
__global__ void k_transpose(const float* __restrict__ s, unsigned short* __restrict__ d,
                            int rows, int cols) {       // d[c*rows+r] = s[r*cols+c]
  long i = blockIdx.x * (long)blockDim.x + threadIdx.x;
  if (i < (long)rows * cols) {
    int r = (int)(i / cols), c = (int)(i % cols);
    d[(size_t)c * rows + r] = f2bf(s[i]);
  }
}
__global__ void k_zeros(float* p, int n) {
  int i = blockIdx.x * blockDim.x + threadIdx.x;
  if (i < n) p[i] = 0.f;
}
__global__ void k_masks(const unsigned char* __restrict__ missing, float* __restrict__ mrow) {
  int n = blockIdx.x * blockDim.x + threadIdx.x;
  if (n < NB) {
    mrow[0 * NB + n] = 1.f;
    mrow[1 * NB + n] = 1.f;
    mrow[2 * NB + n] = missing[n * 2 + 0] ? 1.f : 0.f;
    mrow[3 * NB + n] = missing[n * 2 + 1] ? 1.f : 0.f;
  }
}
__global__ void k_sumsq(const float* __restrict__ x, long n, float* __restrict__ slot) {
  long i0 = blockIdx.x * (long)blockDim.x + threadIdx.x;
  float s = 0.f;
  for (long i = i0; i < n; i += (long)gridDim.x * blockDim.x) { float v = x[i]; s += v * v; }
#pragma unroll
  for (int off = 16; off > 0; off >>= 1) s += __shfl_xor(s, off, 32);
  if ((threadIdx.x & 31) == 0) atomicAdd(slot, s);
}

// Column-wise batchnorm over NB rows, bf16 out; optional ReLU on the input.
__global__ __launch_bounds__(256) void k_bn(
    const float* __restrict__ x, const float* __restrict__ gamma,
    const float* __restrict__ beta, unsigned short* __restrict__ out, int relu_in)
{
  __shared__ float sm[256];
  const int c = blockIdx.x, tid = threadIdx.x;
  float s = 0.f, s2 = 0.f;
  for (int n = tid; n < NB; n += 256) {
    float v = x[(size_t)n * HDm + c];
    if (relu_in) v = fmaxf(v, 0.f);
    s += v; s2 += v * v;
  }
  sm[tid] = s; __syncthreads();
  for (int st = 128; st > 0; st >>= 1) { if (tid < st) sm[tid] += sm[tid + st]; __syncthreads(); }
  float mean = sm[0] / NB; __syncthreads();
  sm[tid] = s2; __syncthreads();
  for (int st = 128; st > 0; st >>= 1) { if (tid < st) sm[tid] += sm[tid + st]; __syncthreads(); }
  float var = sm[0] / NB - mean * mean;
  float inv = rsqrtf(var + EPSN);
  float g = gamma[c], be = beta[c];
  for (int n = tid; n < NB; n += 256) {
    float v = x[(size_t)n * HDm + c];
    if (relu_in) v = fmaxf(v, 0.f);
    out[(size_t)n * HDm + c] = f2bf((v - mean) * inv * g + be);
  }
}
__global__ void k_rowsq(const unsigned short* __restrict__ bn, float* __restrict__ sq) {
  int n = blockIdx.x * blockDim.x + threadIdx.x;
  if (n < NB) {
    float s = 0.f;
    for (int c = 0; c < HDm; ++c) { float v = bf2f(bn[(size_t)n * HDm + c]); s += v * v; }
    sq[n] = s;
  }
}

// Fused multi-bandwidth Gaussian kernels + eps mix + mask + accumulate.
__global__ void k_sim(const float* __restrict__ DW, const float* __restrict__ DG,
                      const float* __restrict__ sW, const float* __restrict__ sG,
                      const float* __restrict__ epsp, const float* __restrict__ mr,
                      float* __restrict__ SS, int init)
{
  const float denom = (float)NB * (float)NB - (float)NB;
  const float bwW = sW[0] / denom * 0.5f;   // /= GK_MUL**(GK_NUM//2) = 2
  const float bwG = sG[0] / denom * 0.5f;
  const float iW = 1.f / bwW, iG = 1.f / bwG;
  const float e = 1.f / (1.f + expf(-epsp[0]));
  const size_t tot = (size_t)NB * NB;
  for (size_t i = blockIdx.x * (size_t)blockDim.x + threadIdx.x; i < tot;
       i += (size_t)gridDim.x * blockDim.x) {
    int r = (int)(i / NB), c = (int)(i % NB);
    float dW = DW[i], dG = DG[i];
    float kW = (expf(-dW * iW) + expf(-dW * iW * 0.5f) + expf(-dW * iW * 0.25f)) * (1.f / 3.f);
    float kG = (expf(-dG * iG) + expf(-dG * iG * 0.5f) + expf(-dG * iG * 0.25f)) * (1.f / 3.f);
    float v = ((1.f - e) * kW + e) * kG * mr[r] * mr[c];
    SS[i] = init ? v : SS[i] + v;
  }
}
// agg = threshold(sim_sum / max(denom,1)) -> bf16 (A operand of GCN GEMMs)
__global__ void k_agg(const float* __restrict__ SS, const float* __restrict__ m2,
                      const float* __restrict__ m3, const float* __restrict__ thrp,
                      unsigned short* __restrict__ agg)
{
  const float thr = 1.f / (1.f + expf(-thrp[0]));
  const size_t tot = (size_t)NB * NB;
  for (size_t i = blockIdx.x * (size_t)blockDim.x + threadIdx.x; i < tot;
       i += (size_t)gridDim.x * blockDim.x) {
    int r = (int)(i / NB), c = (int)(i % NB);
    float den = 2.f + m2[r] * m2[c] + m3[r] * m3[c];
    float a = SS[i] / fmaxf(den, 1.f);
    agg[i] = f2bf(a > thr ? a : 0.f);
  }
}

// Assemble z[N,4,H]: imputation (gate collapses to identity in the source) +
// type embeddings; writes f32 and bf16 token stream.
__global__ void k_assemble(const float* x0, const float* x1, const float* x2, const float* x3,
                           const float* aux0, const float* aux1, const float* mrow,
                           const float* te, float* Z, unsigned short* Zb)
{
  long i = blockIdx.x * (long)blockDim.x + threadIdx.x;
  if (i >= (long)NB * HDm) return;
  int n = (int)(i / HDm), c = (int)(i % HDm);
  float m2 = mrow[2 * NB + n], m3 = mrow[3 * NB + n];
  float e0 = x0[i] + te[c];
  float e1 = x1[i] + te[HDm + c];
  float e2 = (m2 > 0.f ? x2[i] : aux0[i]) + te[2 * HDm + c];
  float e3 = (m3 > 0.f ? x3[i] : aux1[i]) + te[3 * HDm + c];
  size_t b = (size_t)n * 4 * HDm + c;
  Z[b] = e0; Z[b + HDm] = e1; Z[b + 2 * HDm] = e2; Z[b + 3 * HDm] = e3;
  Zb[b] = f2bf(e0); Zb[b + HDm] = f2bf(e1); Zb[b + 2 * HDm] = f2bf(e2); Zb[b + 3 * HDm] = f2bf(e3);
}

// Row LayerNorm (H=256, 1 block/row) with optional residual and trailing ReLU.
__global__ __launch_bounds__(256) void k_ln(
    const float* __restrict__ x, const float* __restrict__ resid,
    const float* __restrict__ gamma, const float* __restrict__ beta,
    float* __restrict__ outf, unsigned short* __restrict__ outb, int post_relu)
{
  __shared__ float sm[256];
  const int row = blockIdx.x, tid = threadIdx.x;
  float v = x[(size_t)row * HDm + tid] + (resid ? resid[(size_t)row * HDm + tid] : 0.f);
  sm[tid] = v; __syncthreads();
  for (int st = 128; st > 0; st >>= 1) { if (tid < st) sm[tid] += sm[tid + st]; __syncthreads(); }
  float mean = sm[0] / HDm; __syncthreads();
  float d = v - mean;
  sm[tid] = d * d; __syncthreads();
  for (int st = 128; st > 0; st >>= 1) { if (tid < st) sm[tid] += sm[tid + st]; __syncthreads(); }
  float var = sm[0] / HDm;
  float y = d * rsqrtf(var + EPSN) * gamma[tid] + beta[tid];
  if (post_relu) y = fmaxf(y, 0.f);
  outf[(size_t)row * HDm + tid] = y;
  outb[(size_t)row * HDm + tid] = f2bf(y);
}

// Masked MHA over M=4 modality tokens; one thread per (sample, head).
__global__ void k_attn(const float* __restrict__ Q, const float* __restrict__ Kt,
                       const float* __restrict__ V, const float* __restrict__ mrow,
                       float* __restrict__ O, int heads)
{
  int idx = blockIdx.x * blockDim.x + threadIdx.x;
  if (idx >= NB * heads) return;
  int b = idx / heads, h = idx % heads;
  int dh = HDm / heads;
  float scale = rsqrtf((float)dh);
  float msk[4];
  for (int t = 0; t < 4; ++t) msk[t] = mrow[t * NB + b];
  for (int qt = 0; qt < 4; ++qt) {
    const float* qp = Q + ((size_t)(b * 4 + qt)) * HDm + h * dh;
    float s[4];
    for (int t = 0; t < 4; ++t) {
      const float* kp = Kt + ((size_t)(b * 4 + t)) * HDm + h * dh;
      float acc = 0.f;
      for (int d = 0; d < dh; ++d) acc += qp[d] * kp[d];
      s[t] = (msk[t] > 0.f) ? acc * scale : -3.402823466e38f;
    }
    float mx = fmaxf(fmaxf(s[0], s[1]), fmaxf(s[2], s[3]));
    float p[4], ps = 0.f;
    for (int t = 0; t < 4; ++t) { p[t] = expf(s[t] - mx); ps += p[t]; }
    float ips = 1.f / ps;
    for (int t = 0; t < 4; ++t) p[t] *= ips;
    float* op = O + ((size_t)(b * 4 + qt)) * HDm + h * dh;
    for (int d = 0; d < dh; ++d) {
      float o = 0.f;
      for (int t = 0; t < 4; ++t) o += p[t] * V[((size_t)(b * 4 + t)) * HDm + h * dh + d];
      op[d] = o;
    }
  }
}

__global__ void k_fc2(const float* __restrict__ y1, const float* __restrict__ w,
                      const float* __restrict__ b, float* __restrict__ y)
{
  int n = blockIdx.x * blockDim.x + threadIdx.x;
  if (n < NB) {
    float acc = b[0];
    for (int k = 0; k < 512; ++k) acc += y1[(size_t)n * 512 + k] * w[k];
    y[n] = acc;
  }
}
__global__ void k_lstab(const float* __restrict__ scal, float* __restrict__ out) {
  if (threadIdx.x == 0 && blockIdx.x == 0) {
    float l = 0.f;
    for (int i = 0; i < 4; ++i) l += fabsf(sqrtf(scal[i]) - sqrtf(scal[4 + i]));
    out[NB] = l;
  }
}

// ---------------------------------------------------------------------------
// Workspace layout (bytes). Phases overlay dead regions.
// ---------------------------------------------------------------------------
constexpr size_t MBy = 1024ull * 1024ull;
constexpr size_t OFF_SIMSUM = 0;                       // 64MB f32 NxN   (sim phase)
constexpr size_t OFF_DW     = 64 * MBy;                // 64MB f32 NxN   (sim phase)
constexpr size_t OFF_DG     = 128 * MBy;               // 64MB f32 NxN   (sim phase)
constexpr size_t OFF_AGG    = OFF_DW;                  // 32MB bf16 NxN  (GCN phase)
constexpr size_t OFF_Q      = 0;                       // 16MB  (tran phase, SIMSUM dead)
constexpr size_t OFF_KB     = 16 * MBy;                // 16MB
constexpr size_t OFF_V      = 32 * MBy;                // 16MB
constexpr size_t OFF_ATTO   = 48 * MBy;                // 16MB
constexpr size_t OFF_TMP    = OFF_Q;                   // reuse after attn
constexpr size_t OFF_TMP2   = OFF_KB;
constexpr size_t OFF_H      = 64 * MBy;                // 16MB (AGG dead by tran phase)
constexpr size_t OFF_HBF    = 80 * MBy;                // 8MB
constexpr size_t OFF_ATTOBF = 88 * MBy;                // 8MB
constexpr size_t OFF_FFBF   = 96 * MBy;                // 32MB bf16 [16384,1024]
constexpr size_t OFF_Z      = 128 * MBy;               // 16MB (DG dead)
constexpr size_t OFF_ZBF    = 144 * MBy;               // 8MB
constexpr size_t OFF_Y1     = 152 * MBy;               // 8MB
constexpr size_t OFF_SPT    = 192 * MBy;               // 12 x 128KB transposed simi_proj
constexpr size_t OFF_GT     = OFF_SPT + 12 * 131072;   // 4 x 128KB graph weights^T
constexpr size_t OFF_ATTNT  = OFF_GT + 4 * 131072;     // 8 x 128KB attn weights^T
constexpr size_t OFF_FF1T   = OFF_ATTNT + 8 * 131072;  // 2 x 512KB
constexpr size_t OFF_FF2T   = OFF_FF1T + 2 * 524288;   // 2 x 512KB
constexpr size_t OFF_FC1T   = OFF_FF2T + 2 * 524288;   // 1MB
constexpr size_t OFF_XBF    = OFF_FC1T + 1048576;      // 4 x 2MB
constexpr size_t OFF_PBFA   = OFF_XBF + 4 * 2097152;   // 2MB
constexpr size_t OFF_PBFB   = OFF_PBFA + 2097152;      // 2MB
constexpr size_t OFF_P3     = OFF_PBFB + 2097152;      // 4MB f32
constexpr size_t OFF_BNW    = OFF_P3 + 4194304;        // 2MB
constexpr size_t OFF_BNG    = OFF_BNW + 2097152;       // 2MB
constexpr size_t OFF_SQW    = OFF_BNG + 2097152;       // 16KB
constexpr size_t OFF_SQG    = OFF_SQW + 16384;         // 16KB
constexpr size_t OFF_MROW   = OFF_SQG + 16384;         // 64KB  [4][N]
constexpr size_t OFF_SCAL   = OFF_MROW + 65536;        // 4KB scalars
constexpr size_t OFF_XWBT   = OFF_SCAL + 4096;         // 2MB bf16 [256,4096]
constexpr size_t OFF_T1BF   = OFF_XWBT + 2097152;      // 2MB
constexpr size_t OFF_AUX0   = OFF_T1BF + 2097152;      // 4MB f32
constexpr size_t OFF_AUX1   = OFF_AUX0 + 4194304;      // 4MB f32

static inline unsigned gemm_blocks(int Md, int Nd) {
  long tiles = (long)(Md / 32) * (Nd / 32);            // 32x32 tiles, 8 waves/block
  return (unsigned)((tiles + 7) / 8);
}

extern "C" void kernel_launch(void* const* d_in, const int* in_sizes, int n_in,
                              void* d_out, int out_size, void* d_ws, size_t ws_size,
                              hipStream_t stream)
{
  (void)in_sizes; (void)n_in; (void)out_size; (void)ws_size;
  char* ws = (char*)d_ws;
  auto F = [&](int i) { return (const float*)d_in[i]; };

  const float* X[4] = { F(0), F(1), F(2), F(3) };
  const unsigned char* missing = (const unsigned char*)d_in[4];
  // pytree (sorted) param leaves:
  const float* dissim   = F(9);
  const float* g_b[2][2] = { { F(10), F(12) }, { F(14), F(16) } };
  const float* g_w[2][2] = { { F(11), F(13) }, { F(15), F(17) } };
  const float* fc1b = F(18); const float* fc1w = F(19);
  const float* fc2b = F(20); const float* fc2w = F(21);
  const float* bnbeta = F(22); const float* bngamma = F(23);
  const float* simi_eps = F(24);
  const float* sp_b[4][3]; const float* sp_w[4][3];
  for (int i = 0; i < 4; ++i)
    for (int j = 0; j < 3; ++j) {
      sp_b[i][j] = F(25 + i * 6 + j * 2);
      sp_w[i][j] = F(26 + i * 6 + j * 2);
    }
  const float *wk[2], *wo[2], *wq[2], *wv[2], *ff1b[2], *ff1w[2], *ff2b[2], *ff2w[2];
  const float *ln1b[2], *ln1g[2], *ln2b[2], *ln2g[2];
  for (int t = 0; t < 2; ++t) {
    int b = 49 + t * 12;
    wk[t] = F(b); wo[t] = F(b + 1); wq[t] = F(b + 2); wv[t] = F(b + 3);
    ff1b[t] = F(b + 4); ff1w[t] = F(b + 5); ff2b[t] = F(b + 6); ff2w[t] = F(b + 7);
    ln1b[t] = F(b + 8); ln1g[t] = F(b + 9); ln2b[t] = F(b + 10); ln2g[t] = F(b + 11);
  }
  const float* type_emb = F(73);
  float* OUT = (float*)d_out;

  // workspace pointers
  float* SIMSUM = (float*)(ws + OFF_SIMSUM);
  float* DW = (float*)(ws + OFF_DW);
  float* DG = (float*)(ws + OFF_DG);
  unsigned short* AGG = (unsigned short*)(ws + OFF_AGG);
  float* Qb = (float*)(ws + OFF_Q);  float* Kb = (float*)(ws + OFF_KB);
  float* Vb = (float*)(ws + OFF_V);  float* ATTO = (float*)(ws + OFF_ATTO);
  float* TMP = (float*)(ws + OFF_TMP); float* TMP2 = (float*)(ws + OFF_TMP2);
  float* Hb = (float*)(ws + OFF_H);
  unsigned short* HBF = (unsigned short*)(ws + OFF_HBF);
  unsigned short* ATTOBF = (unsigned short*)(ws + OFF_ATTOBF);
  unsigned short* FFBF = (unsigned short*)(ws + OFF_FFBF);
  float* Z = (float*)(ws + OFF_Z);
  unsigned short* ZBF = (unsigned short*)(ws + OFF_ZBF);
  float* Y1 = (float*)(ws + OFF_Y1);
  unsigned short* SPT = (unsigned short*)(ws + OFF_SPT);
  unsigned short* GT = (unsigned short*)(ws + OFF_GT);
  unsigned short* ATTNT = (unsigned short*)(ws + OFF_ATTNT);
  unsigned short* FF1T = (unsigned short*)(ws + OFF_FF1T);
  unsigned short* FF2T = (unsigned short*)(ws + OFF_FF2T);
  unsigned short* FC1T = (unsigned short*)(ws + OFF_FC1T);
  unsigned short* XBF = (unsigned short*)(ws + OFF_XBF);
  unsigned short* PBFA = (unsigned short*)(ws + OFF_PBFA);
  unsigned short* PBFB = (unsigned short*)(ws + OFF_PBFB);
  float* P3 = (float*)(ws + OFF_P3);
  unsigned short* BNW = (unsigned short*)(ws + OFF_BNW);
  unsigned short* BNG = (unsigned short*)(ws + OFF_BNG);
  float* SQW = (float*)(ws + OFF_SQW);
  float* SQG = (float*)(ws + OFF_SQG);
  float* MROW = (float*)(ws + OFF_MROW);
  float* SCAL = (float*)(ws + OFF_SCAL); // [0..3] |proj|^2, [4..7] |x|^2, [8,9] bw sums
  unsigned short* XWBT = (unsigned short*)(ws + OFF_XWBT);
  unsigned short* T1BF = (unsigned short*)(ws + OFF_T1BF);
  float* AUX[2] = { (float*)(ws + OFF_AUX0), (float*)(ws + OFF_AUX1) };

  const int HH = HDm * HDm;
  const int R16 = NB * 4;             // token rows for transformer

  // ---- setup: scalars, masks, bf16 inputs, transposed bf16 weights --------
  k_zeros<<<1, 64, 0, stream>>>(SCAL, 16);
  k_masks<<<(NB + 255) / 256, 256, 0, stream>>>(missing, MROW);
  for (int i = 0; i < 4; ++i)
    k_cvt<<<(NB * HDm + 255) / 256, 256, 0, stream>>>(X[i], XBF + (size_t)i * NB * HDm, (long)NB * HDm);
  for (int i = 0; i < 4; ++i)
    for (int j = 0; j < 3; ++j)
      k_transpose<<<(HH + 255) / 256, 256, 0, stream>>>(sp_w[i][j], SPT + (size_t)(i * 3 + j) * HH, HDm, HDm);
  for (int mi = 0; mi < 2; ++mi)
    for (int l = 0; l < 2; ++l)
      k_transpose<<<(HH + 255) / 256, 256, 0, stream>>>(g_w[mi][l], GT + (size_t)(mi * 2 + l) * HH, HDm, HDm);
  for (int t = 0; t < 2; ++t) {
    k_transpose<<<(HH + 255) / 256, 256, 0, stream>>>(wq[t], ATTNT + (size_t)(t * 4 + 0) * HH, HDm, HDm);
    k_transpose<<<(HH + 255) / 256, 256, 0, stream>>>(wk[t], ATTNT + (size_t)(t * 4 + 1) * HH, HDm, HDm);
    k_transpose<<<(HH + 255) / 256, 256, 0, stream>>>(wv[t], ATTNT + (size_t)(t * 4 + 2) * HH, HDm, HDm);
    k_transpose<<<(HH + 255) / 256, 256, 0, stream>>>(wo[t], ATTNT + (size_t)(t * 4 + 3) * HH, HDm, HDm);
    k_transpose<<<(4 * HH + 255) / 256, 256, 0, stream>>>(ff1w[t], FF1T + (size_t)t * 4 * HH, HDm, 4 * HDm);
    k_transpose<<<(4 * HH + 255) / 256, 256, 0, stream>>>(ff2w[t], FF2T + (size_t)t * 4 * HH, 4 * HDm, HDm);
  }
  k_transpose<<<(1024 * 512 + 255) / 256, 256, 0, stream>>>(fc1w, FC1T, 1024, 512);

  // ---- similarity graph: projections, batchnorm, WMMA Gram+dist, sim ------
  for (int i = 0; i < 4; ++i) {
    const unsigned short* xb = XBF + (size_t)i * NB * HDm;
    k_gemm<true, true, false, true, false><<<gemm_blocks(NB, HDm), 256, 0, stream>>>(
        xb, SPT + (size_t)(i * 3 + 0) * HH, sp_b[i][0], nullptr, PBFA, nullptr, NB, HDm, HDm);
    k_gemm<true, true, false, true, false><<<gemm_blocks(NB, HDm), 256, 0, stream>>>(
        PBFA, SPT + (size_t)(i * 3 + 1) * HH, sp_b[i][1], nullptr, PBFB, nullptr, NB, HDm, HDm);
    k_gemm<true, false, true, false, false><<<gemm_blocks(NB, HDm), 256, 0, stream>>>(
        PBFB, SPT + (size_t)(i * 3 + 2) * HH, sp_b[i][2], P3, nullptr, nullptr, NB, HDm, HDm);
    k_sumsq<<<256, 256, 0, stream>>>(P3, (long)NB * HDm, SCAL + i);       // |proj|^2
    k_sumsq<<<256, 256, 0, stream>>>(X[i], (long)NB * HDm, SCAL + 4 + i); // |x|^2
    k_bn<<<HDm, 256, 0, stream>>>(P3, bngamma, bnbeta, BNW, 1);
    k_bn<<<HDm, 256, 0, stream>>>(X[i], bngamma, bnbeta, BNG, 0);
    k_rowsq<<<(NB + 255) / 256, 256, 0, stream>>>(BNW, SQW);
    k_rowsq<<<(NB + 255) / 256, 256, 0, stream>>>(BNG, SQG);
    k_zeros<<<1, 32, 0, stream>>>(SCAL + 8, 2);
    k_dist<<<gemm_blocks(NB, NB), 256, 0, stream>>>(BNW, SQW, DW, SCAL + 8, HDm);
    k_dist<<<gemm_blocks(NB, NB), 256, 0, stream>>>(BNG, SQG, DG, SCAL + 9, HDm);
    k_sim<<<8192, 256, 0, stream>>>(DW, DG, SCAL + 8, SCAL + 9, simi_eps + i,
                                    MROW + (size_t)i * NB, SIMSUM, i == 0 ? 1 : 0);
  }
  k_agg<<<8192, 256, 0, stream>>>(SIMSUM, MROW + 2 * NB, MROW + 3 * NB, dissim, AGG);

  // ---- GCN imputation for missing modalities (i = 2, 3) -------------------
  for (int mi = 0; mi < 2; ++mi) {
    const unsigned short* xb = XBF + (size_t)(2 + mi) * NB * HDm;
    k_gemm<true, false, false, false, true><<<gemm_blocks(NB, HDm), 256, 0, stream>>>(
        xb, GT + (size_t)(mi * 2 + 0) * HH, g_b[mi][0], nullptr, nullptr, XWBT, NB, HDm, HDm);
    k_gemm<false, true, false, true, false><<<gemm_blocks(NB, HDm), 256, 0, stream>>>(
        AGG, XWBT, nullptr, nullptr, T1BF, nullptr, NB, HDm, NB);
    k_gemm<true, false, false, false, true><<<gemm_blocks(NB, HDm), 256, 0, stream>>>(
        T1BF, GT + (size_t)(mi * 2 + 1) * HH, g_b[mi][1], nullptr, nullptr, XWBT, NB, HDm, HDm);
    k_gemm<false, true, true, false, false><<<gemm_blocks(NB, HDm), 256, 0, stream>>>(
        AGG, XWBT, nullptr, AUX[mi], nullptr, nullptr, NB, HDm, NB);
  }

  // ---- token stream + transformer blocks ----------------------------------
  k_assemble<<<(NB * HDm + 255) / 256, 256, 0, stream>>>(X[0], X[1], X[2], X[3],
      AUX[0], AUX[1], MROW, type_emb, Z, ZBF);

  for (int t = 0; t < 2; ++t) {
    const int heads = (t == 0) ? 4 : 1;
    const unsigned short* WQt = ATTNT + (size_t)(t * 4 + 0) * HH;
    const unsigned short* WKt = ATTNT + (size_t)(t * 4 + 1) * HH;
    const unsigned short* WVt = ATTNT + (size_t)(t * 4 + 2) * HH;
    const unsigned short* WOt = ATTNT + (size_t)(t * 4 + 3) * HH;
    k_gemm<false, false, true, false, false><<<gemm_blocks(R16, HDm), 256, 0, stream>>>(
        ZBF, WQt, nullptr, Qb, nullptr, nullptr, R16, HDm, HDm);
    k_gemm<false, false, true, false, false><<<gemm_blocks(R16, HDm), 256, 0, stream>>>(
        ZBF, WKt, nullptr, Kb, nullptr, nullptr, R16, HDm, HDm);
    k_gemm<false, false, true, false, false><<<gemm_blocks(R16, HDm), 256, 0, stream>>>(
        ZBF, WVt, nullptr, Vb, nullptr, nullptr, R16, HDm, HDm);
    k_attn<<<(NB * heads + 255) / 256, 256, 0, stream>>>(Qb, Kb, Vb, MROW, ATTO, heads);
    k_cvt<<<(R16 * HDm + 255) / 256, 256, 0, stream>>>(ATTO, ATTOBF, (long)R16 * HDm);
    k_gemm<false, false, true, false, false><<<gemm_blocks(R16, HDm), 256, 0, stream>>>(
        ATTOBF, WOt, nullptr, TMP, nullptr, nullptr, R16, HDm, HDm);
    k_ln<<<R16, 256, 0, stream>>>(TMP, Z, ln1g[t], ln1b[t], Hb, HBF, 0);
    k_gemm<true, true, false, true, false><<<gemm_blocks(R16, 4 * HDm), 256, 0, stream>>>(
        HBF, FF1T + (size_t)t * 4 * HH, ff1b[t], nullptr, FFBF, nullptr, R16, 4 * HDm, HDm);
    k_gemm<true, false, true, false, false><<<gemm_blocks(R16, HDm), 256, 0, stream>>>(
        FFBF, FF2T + (size_t)t * 4 * HH, ff2b[t], TMP2, nullptr, nullptr, R16, HDm, 4 * HDm);
    k_ln<<<R16, 256, 0, stream>>>(TMP2, Hb, ln2g[t], ln2b[t], Z, ZBF, 1); // z = relu(block)
  }

  // ---- output head: comb = z.reshape(N,1024) ------------------------------
  k_gemm<true, true, true, false, false><<<gemm_blocks(NB, 512), 256, 0, stream>>>(
      ZBF, FC1T, fc1b, Y1, nullptr, nullptr, NB, 512, 1024);
  k_fc2<<<(NB + 255) / 256, 256, 0, stream>>>(Y1, fc2w, fc2b, OUT);
  k_lstab<<<1, 32, 0, stream>>>(SCAL, OUT);
}
